// ContrastiveLoss_43267500540206
// MI455X (gfx1250) — matrix operand order repaired and checked by reading, not compiled, not run on hardware
//
#include <hip/hip_runtime.h>

typedef _Float16 v16h __attribute__((ext_vector_type(16)));
typedef _Float16 v8h  __attribute__((ext_vector_type(8)));
typedef float    v8f  __attribute__((ext_vector_type(8)));
typedef float    v4f  __attribute__((ext_vector_type(4)));

#define BB      4096           // batch B
#define DD      256            // feature dim
#define NROWS   8192           // 2B
#define NCTILE  64             // 8192 / 128 column tiles
#define INV_T   14.285714285714286f

// ---------------- Kernel A: L2-normalize rows, emit f16 reps ----------------
// grid: NROWS/8 blocks x 256 threads; one wave32 per row, 8 floats per lane.
__global__ __launch_bounds__(256) void nrm_kernel(const float* __restrict__ zi,
                                                  const float* __restrict__ zj,
                                                  _Float16* __restrict__ reps) {
    const int w    = threadIdx.x >> 5;
    const int lane = threadIdx.x & 31;
    const int row  = blockIdx.x * 8 + w;
    const float* src = (row < BB) ? (zi + (size_t)row * DD)
                                  : (zj + (size_t)(row - BB) * DD);
    v4f a = *(const v4f*)(src + lane * 8);
    v4f b = *(const v4f*)(src + lane * 8 + 4);
    float ss = a.x*a.x + a.y*a.y + a.z*a.z + a.w*a.w
             + b.x*b.x + b.y*b.y + b.z*b.z + b.w*b.w;
    #pragma unroll
    for (int off = 16; off; off >>= 1) ss += __shfl_xor(ss, off, 32);
    const float s = 1.0f / fmaxf(sqrtf(ss), 1e-12f);
    v8h o;
    o[0] = (_Float16)(a.x * s); o[1] = (_Float16)(a.y * s);
    o[2] = (_Float16)(a.z * s); o[3] = (_Float16)(a.w * s);
    o[4] = (_Float16)(b.x * s); o[5] = (_Float16)(b.y * s);
    o[6] = (_Float16)(b.z * s); o[7] = (_Float16)(b.w * s);
    *(v8h*)(reps + (size_t)row * DD + lane * 8) = o;
}

// ---------------- Kernel P: positive-pair logits ----------------
// pos[r] = invT * <reps[r], reps[r ^ 4096]>  (f16 inputs, f32 accumulate,
// matching the WMMA GEMM numerics). One wave32 per row.
__global__ __launch_bounds__(256) void pos_kernel(const _Float16* __restrict__ reps,
                                                  float* __restrict__ pos) {
    const int w    = threadIdx.x >> 5;
    const int lane = threadIdx.x & 31;
    const int row  = blockIdx.x * 8 + w;
    const int par  = row ^ BB;                 // (i, i+B) and (i+B, i)
    v8h a = *(const v8h*)(reps + (size_t)row * DD + lane * 8);
    v8h b = *(const v8h*)(reps + (size_t)par * DD + lane * 8);
    float d = 0.0f;
    #pragma unroll
    for (int e = 0; e < 8; ++e) d = fmaf((float)a[e], (float)b[e], d);
    #pragma unroll
    for (int off = 16; off; off >>= 1) d += __shfl_xor(d, off, 32);
    if (lane == 0) pos[row] = d * INV_T;
}

// ---------------- Kernel B: WMMA sim tiles + exp row partials ----------------
// grid: 64x64 blocks of 256 threads. Block tile = 128x128 of sim.
// Wave w computes rows [rowBase+w*16, +16) against 8 column tiles of 16.
__global__ __launch_bounds__(256) void sim_kernel(const _Float16* __restrict__ reps,
                                                  float* __restrict__ partial) {
    const int lane    = threadIdx.x & 31;
    const int w       = threadIdx.x >> 5;
    const int l16     = lane & 15;
    const int half    = lane >> 4;                 // 0: lanes 0-15, 1: lanes 16-31
    const int rowBase = blockIdx.y * 128 + w * 16;
    const int colBase = blockIdx.x * 128;

    // ---- Preload A fragments for all 8 K-steps (K = 256 = 8 x 32) ----
    // A 16x32 f16 layout (ISA 7.12.2): lane row M = l16; VGPR0-3 hold
    // K = half*8 .. +7 (packed pairs), VGPR4-7 hold K = 16 + half*8 .. +7.
    // => two contiguous 8-half (16B) loads per fragment; reused for 8 N-tiles.
    const _Float16* arow = reps + (size_t)(rowBase + l16) * DD;
    v16h afrag[8];
    #pragma unroll
    for (int k = 0; k < 8; ++k) {
        const _Float16* p = arow + k * 32 + half * 8;
        v8h lo = *(const v8h*)(p);
        v8h hi = *(const v8h*)(p + 16);
        afrag[k] = __builtin_shufflevector(lo, hi,
                     0,1,2,3,4,5,6,7,8,9,10,11,12,13,14,15);
    }

    float racc[8];                                 // per-lane column partials
    #pragma unroll
    for (int v = 0; v < 8; ++v) racc[v] = 0.0f;

    for (int nt = 0; nt < 8; ++nt) {
        const int gcol = colBase + nt * 16 + l16;      // this lane's N column
        // B 32x16 f16 layout: lane column N = l16; VGPR0-3 hold K = half*8 .. +7,
        // VGPR4-7 hold K = 16 + half*8 .. +7 (two 16B loads, same as A).
        // Issue all 16 loads first so they clause and overlap prior VALU work.
        const _Float16* brow = reps + (size_t)gcol * DD;   // B = reps^T
        v16h bfrag[8];
        #pragma unroll
        for (int k = 0; k < 8; ++k) {
            const _Float16* p = brow + k * 32 + half * 8;
            v8h lo = *(const v8h*)(p);
            v8h hi = *(const v8h*)(p + 16);
            bfrag[k] = __builtin_shufflevector(lo, hi,
                         0,1,2,3,4,5,6,7,8,9,10,11,12,13,14,15);
        }
        v8f c = {};
        #pragma unroll
        for (int k = 0; k < 8; ++k)
            c = __builtin_amdgcn_wmma_f32_16x16x32_f16(
                    false, afrag[k], false, bfrag[k], (short)0, c, false, false);

        // D layout: lanes 0-15 hold M = v (VGPR v), lanes 16-31 hold M = 8+v.
        // Diagonal term: exp(-1e9/T) underflows to 0 in fp32 -> drop it.
        // No cross-lane work here: accumulate per-lane, reduce once at the end.
        #pragma unroll
        for (int v = 0; v < 8; ++v) {
            const int grow = rowBase + v + half * 8;
            float e = __expf(c[v] * INV_T);
            e = (grow == gcol) ? 0.0f : e;
            racc[v] += e;
        }
    }

    // Single 16-lane-group butterfly per row (8 x 4 shuffles per wave).
    #pragma unroll
    for (int v = 0; v < 8; ++v) {
        float e = racc[v];
        e += __shfl_xor(e, 1, 32);
        e += __shfl_xor(e, 2, 32);
        e += __shfl_xor(e, 4, 32);
        e += __shfl_xor(e, 8, 32);
        racc[v] = e;
    }
    if (l16 == 0) {
        #pragma unroll
        for (int v = 0; v < 8; ++v) {
            const int grow = rowBase + v + half * 8;
            partial[(size_t)grow * NCTILE + blockIdx.x] = racc[v];
        }
    }
}

// ---------------- Kernel C: deterministic final reduction ----------------
__global__ __launch_bounds__(256) void loss_kernel(const float* __restrict__ partial,
                                                   const float* __restrict__ pos,
                                                   float* __restrict__ out) {
    __shared__ float red[256];
    float acc = 0.0f;
    for (int r = threadIdx.x; r < NROWS; r += 256) {
        const float* p = partial + (size_t)r * NCTILE;
        float s = 0.0f;
        #pragma unroll
        for (int k = 0; k < NCTILE; ++k) s += p[k];
        acc += logf(s) - pos[r];                  // loss_i = log(denom) - pos_i
    }
    red[threadIdx.x] = acc;
    __syncthreads();
    #pragma unroll
    for (int st = 128; st; st >>= 1) {
        if (threadIdx.x < st) red[threadIdx.x] += red[threadIdx.x + st];
        __syncthreads();
    }
    if (threadIdx.x == 0) out[0] = red[0] / (float)NROWS;
}

extern "C" void kernel_launch(void* const* d_in, const int* in_sizes, int n_in,
                              void* d_out, int out_size, void* d_ws, size_t ws_size,
                              hipStream_t stream) {
    const float* zi = (const float*)d_in[0];
    const float* zj = (const float*)d_in[1];
    float* out = (float*)d_out;

    char* ws = (char*)d_ws;
    _Float16* reps = (_Float16*)ws;                                         // 4 MB
    float* partial = (float*)(ws + (size_t)NROWS * DD * sizeof(_Float16));  // 2 MB
    float* pos     = (float*)((char*)partial + (size_t)NROWS * NCTILE * sizeof(float)); // 32 KB

    nrm_kernel<<<NROWS / 8, 256, 0, stream>>>(zi, zj, reps);
    pos_kernel<<<NROWS / 8, 256, 0, stream>>>(reps, pos);
    sim_kernel<<<dim3(NCTILE, NCTILE), 256, 0, stream>>>(reps, partial);
    loss_kernel<<<1, 256, 0, stream>>>(partial, pos, out);
}
